// SlotAttentionWriter_86208583565575
// MI455X (gfx1250) — matrix operand order repaired and checked by reading, not compiled, not run on hardware
//
#include <hip/hip_runtime.h>
#include <hip/hip_bf16.h>

// ---------------------------------------------------------------------------
// Slot-Attention writer for MI455X (gfx1250, wave32, WMMA).
// All GEMMs run through v_wmma_f32_16x16x32_f16 (f32 accumulate).
// The big projection GEMMs stage B tiles into LDS with
// global_load_async_to_lds_b128 (ASYNCcnt) double buffering.
// ---------------------------------------------------------------------------

typedef __attribute__((ext_vector_type(16))) _Float16 v16h;
typedef __attribute__((ext_vector_type(8)))  _Float16 v8h;
typedef __attribute__((ext_vector_type(8)))  float    v8f;

#define DD 1024
#define BB 8
#define KSLOT 64
#define MTOK 4096
#define BM (BB * MTOK)   // 32768 total tokens
#define BKR (BB * KSLOT) // 512 total slot rows

union AFrag { v16h v; v8h h[2]; };

// A operand (16 rows x 32 K, f16). Lane layout used here (contiguous K, same order as B):
// lanes 0-15 : row m0+lane, K = k0+{0..15}
// lanes 16-31: row m0+lane-16, K = k0+{16..31}
__device__ __forceinline__ v16h load_a_frag(const _Float16* A, int lda,
                                            int m0, int k0, int lane) {
  const int lr = lane & 15, hi = lane >> 4;
  const _Float16* p = A + (size_t)(m0 + lr) * lda + k0 + hi * 16;
  AFrag f;
  f.h[0] = *(const v8h*)(p);
  f.h[1] = *(const v8h*)(p + 8);
  return f.v;
}

// B operand (32 K x 16 cols, f16) for C = X @ W^T:
// lane n (0-15) holds W[n0+n][k0 .. k0+15]  (contiguous 32B),
// lane n+16     holds W[n0+n][k0+16 .. k0+31].
__device__ __forceinline__ v16h load_b_frag(const _Float16* W, size_t ldw,
                                            int n0, int k0, int lane) {
  const _Float16* p = W + (size_t)(n0 + (lane & 15)) * ldw + k0 + (lane >> 4) * 16;
  return *(const v16h*)(p);
}

// B fragment from a 16B-aligned (not necessarily 32B-aligned) location.
__device__ __forceinline__ v16h load_b_frag16(const _Float16* p) {
  AFrag f;
  f.h[0] = *(const v8h*)(p);
  f.h[1] = *(const v8h*)(p + 8);
  return f.v;
}

__device__ __forceinline__ v8f wmma_f16(v16h a, v16h b, v8f c) {
  return __builtin_amdgcn_wmma_f32_16x16x32_f16(false, a, false, b,
                                                (short)0, c, false, false);
}

__device__ __forceinline__ float wave_sum32(float v) {
#pragma unroll
  for (int off = 16; off > 0; off >>= 1) v += __shfl_xor(v, off, 32);
  return v;
}

// ---------------------------------------------------------------------------
// Elementwise helpers
// ---------------------------------------------------------------------------
__global__ void cvt_f32_to_f16(const float* __restrict__ src,
                               _Float16* __restrict__ dst, int n) {
  int i = blockIdx.x * blockDim.x + threadIdx.x;
  if (i < n) dst[i] = (_Float16)src[i];
}

__global__ void zero_f32(float* __restrict__ p, int n) {
  int i = blockIdx.x * blockDim.x + threadIdx.x;
  if (i < n) p[i] = 0.0f;
}

// rmsnorm over last dim d, one 256-thread block per row.
// Y = f16(x * rsqrt(mean(x^2)+eps) * w); Yraw (optional) = f16(x).
__global__ void __launch_bounds__(256)
rmsnorm_f16(const float* __restrict__ X, const float* __restrict__ w,
            _Float16* __restrict__ Y, _Float16* __restrict__ Yraw, int d) {
  const int row = blockIdx.x;
  const float* x = X + (size_t)row * d;
  float ss = 0.0f;
  for (int i = threadIdx.x; i < d; i += 256) {
    float v = x[i];
    ss += v * v;
  }
  __shared__ float red[8];
  __shared__ float scale_s;
  float s = wave_sum32(ss);
  if ((threadIdx.x & 31) == 0) red[threadIdx.x >> 5] = s;
  __syncthreads();
  if (threadIdx.x == 0) {
    float t = 0.0f;
#pragma unroll
    for (int i = 0; i < 8; ++i) t += red[i];
    scale_s = rsqrtf(t / (float)d + 1e-6f);
  }
  __syncthreads();
  const float scale = scale_s;
  for (int i = threadIdx.x; i < d; i += 256) {
    float v = x[i];
    Y[(size_t)row * d + i] = (_Float16)(v * scale * w[i]);
    if (Yraw) Yraw[(size_t)row * d + i] = (_Float16)v;
  }
}

// ---------------------------------------------------------------------------
// GEMM: C[R x N] = A[R x Kd] @ W[N x Kd]^T, f16 in/out, f32 acc.
// Block tile = 128 rows x 64 cols (8 waves, each 16x64). The shared 64x32
// B tile per K-step is staged into LDS via global_load_async_to_lds_b128
// (double buffered, ASYNCcnt), so all 8 waves reuse one global fetch.
// REQUIREMENT: R % 128 == 0, N % 64 == 0, Kd % 32 == 0 (true for all uses).
// ---------------------------------------------------------------------------
#define BROW 40  // LDS row pitch in halves (80B) -> conflict-free ds_load_b128

__global__ void __launch_bounds__(256)
gemm_nt_f16(const _Float16* __restrict__ A, const _Float16* __restrict__ W,
            _Float16* __restrict__ C, int R, int N, int Kd) {
  __shared__ _Float16 Bs[2][64 * BROW];
  const int t = threadIdx.x;
  const int lane = t & 31;
  const int wave = t >> 5;
  const int tilesN = N >> 6;
  const int bm = blockIdx.x / tilesN;
  const int bn = blockIdx.x - bm * tilesN;
  const int m0 = bm * 128 + wave * 16;
  const int n0 = bn << 6;

  // Per-thread staging slot: row = t/4 (0..63), 16B chunk = t%4.
  const int srow = t >> 2;
  const int schunk = (t & 3) * 8;  // halves
  const _Float16* gsrc = W + (size_t)(n0 + srow) * Kd + schunk;

  {  // stage K-step 0 into buffer 0
    unsigned lds = (unsigned)(size_t)&Bs[0][srow * BROW + schunk];
    asm volatile("global_load_async_to_lds_b128 %0, %1, off"
                 :: "v"(lds), "v"(gsrc) : "memory");
  }

  v8f acc0 = {}, acc1 = {}, acc2 = {}, acc3 = {};
  const int lr = lane & 15, hi = lane >> 4;

  for (int k0 = 0; k0 < Kd; k0 += 32) {
    const int cur = (k0 >> 5) & 1;
    if (k0 + 32 < Kd) {  // prefetch next B tile into the other buffer
      unsigned lds = (unsigned)(size_t)&Bs[cur ^ 1][srow * BROW + schunk];
      const _Float16* g = gsrc + k0 + 32;
      asm volatile("global_load_async_to_lds_b128 %0, %1, off"
                   :: "v"(lds), "v"(g) : "memory");
      asm volatile("s_wait_asynccnt 0x1" ::: "memory");  // cur buffer landed
    } else {
      asm volatile("s_wait_asynccnt 0x0" ::: "memory");
    }
    __syncthreads();

    v16h a = load_a_frag(A, Kd, m0, k0, lane);
    const _Float16* bbase = &Bs[cur][0] + hi * 16;
    v16h b0 = load_b_frag16(bbase + (lr)      * BROW);
    v16h b1 = load_b_frag16(bbase + (lr + 16) * BROW);
    v16h b2 = load_b_frag16(bbase + (lr + 32) * BROW);
    v16h b3 = load_b_frag16(bbase + (lr + 48) * BROW);
    acc0 = wmma_f16(a, b0, acc0);
    acc1 = wmma_f16(a, b1, acc1);
    acc2 = wmma_f16(a, b2, acc2);
    acc3 = wmma_f16(a, b3, acc3);
    __syncthreads();  // all waves done with `cur` before it is re-staged
  }

#pragma unroll
  for (int r = 0; r < 8; ++r) {
    size_t row = (size_t)(m0 + r + 8 * hi) * N;
    C[row + n0 + lr]      = (_Float16)acc0[r];
    C[row + n0 + 16 + lr] = (_Float16)acc1[r];
    C[row + n0 + 32 + lr] = (_Float16)acc2[r];
    C[row + n0 + 48 + lr] = (_Float16)acc3[r];
  }
}

// ---------------------------------------------------------------------------
// scores = (q @ k^T) * d^-0.5 ; softmax over the 64-slot axis (all 64 slots
// resident in one wave: 4 slot-tiles x 2 m-tiles of accumulators); then
// partial per-slot row sums (for the second normalization) via atomicAdd.
// ---------------------------------------------------------------------------
__global__ void __launch_bounds__(256)
scores_softmax(const _Float16* __restrict__ q16, const _Float16* __restrict__ k16,
               _Float16* __restrict__ attn, float* __restrict__ rowsum) {
  const int lane = threadIdx.x & 31;
  const int wid = (blockIdx.x << 3) + (threadIdx.x >> 5);
  const int b = wid >> 7;            // 128 m-tiles (of 32) per batch
  const int m0 = (wid & 127) << 5;
  const _Float16* qb = q16 + (size_t)b * KSLOT * DD;
  const _Float16* kb = k16 + (size_t)b * MTOK * DD;
  v8f acc[4][2];
  {
    v8f z = {};
#pragma unroll
    for (int s = 0; s < 4; ++s) { acc[s][0] = z; acc[s][1] = z; }
  }
  for (int k0 = 0; k0 < DD; k0 += 32) {
    v16h b0 = load_b_frag(kb, DD, m0, k0, lane);
    v16h b1 = load_b_frag(kb, DD, m0 + 16, k0, lane);
#pragma unroll
    for (int s = 0; s < 4; ++s) {
      v16h a = load_a_frag(qb, DD, s * 16, k0, lane);
      acc[s][0] = wmma_f16(a, b0, acc[s][0]);
      acc[s][1] = wmma_f16(a, b1, acc[s][1]);
    }
  }
#pragma unroll
  for (int s = 0; s < 4; ++s)
#pragma unroll
    for (int n = 0; n < 2; ++n)
#pragma unroll
      for (int r = 0; r < 8; ++r) acc[s][n][r] *= 0.03125f;  // 1024^-0.5

  const int lr = lane & 15, hi = lane >> 4;
  // softmax across slots: in-lane (4 tiles x 8 regs) + cross-half shuffle
#pragma unroll
  for (int n = 0; n < 2; ++n) {
    float mx = -3.0e38f;
#pragma unroll
    for (int s = 0; s < 4; ++s)
#pragma unroll
      for (int r = 0; r < 8; ++r) mx = fmaxf(mx, acc[s][n][r]);
    mx = fmaxf(mx, __shfl_xor(mx, 16, 32));
    float sum = 0.0f;
#pragma unroll
    for (int s = 0; s < 4; ++s)
#pragma unroll
      for (int r = 0; r < 8; ++r) {
        float e = __expf(acc[s][n][r] - mx);
        acc[s][n][r] = e;
        sum += e;
      }
    sum += __shfl_xor(sum, 16, 32);
    float inv = 1.0f / sum;
#pragma unroll
    for (int s = 0; s < 4; ++s)
#pragma unroll
      for (int r = 0; r < 8; ++r) acc[s][n][r] *= inv;
  }
  // per-slot partial sums over this wave's 32 m columns + attn store
#pragma unroll
  for (int s = 0; s < 4; ++s) {
#pragma unroll
    for (int r = 0; r < 8; ++r) {
      float v = acc[s][0][r] + acc[s][1][r];
#pragma unroll
      for (int off = 1; off < 16; off <<= 1) v += __shfl_xor(v, off, 32);
      const int slot = s * 16 + r + 8 * hi;
      if (lr == 0) atomicAdd(&rowsum[b * KSLOT + slot], v);
      _Float16* dst = attn + (size_t)(b * KSLOT + slot) * MTOK + m0 + lr;
      dst[0]  = (_Float16)acc[s][0][r];
      dst[16] = (_Float16)acc[s][1][r];
    }
  }
}

// ---------------------------------------------------------------------------
// updates[b] = diag(1/(rowsum+eps)) * attn[b] @ V[b]
// V consumed as vT (d x B*M), so B fragments are contiguous loads.
// ---------------------------------------------------------------------------
__global__ void __launch_bounds__(256)
attn_v_gemm(const _Float16* __restrict__ attn, const _Float16* __restrict__ vT,
            const float* __restrict__ rowsum, _Float16* __restrict__ upd) {
  const int lane = threadIdx.x & 31;
  const int wid = (blockIdx.x << 3) + (threadIdx.x >> 5);
  const int b = wid >> 6;            // 4 slot-tiles * 16 col-strips per batch
  const int t = wid & 63;
  const int s0 = (t >> 4) << 4;
  const int n0 = (t & 15) << 6;
  const _Float16* ab = attn + (size_t)b * KSLOT * MTOK;
  const _Float16* vb = vT + (size_t)b * MTOK;  // row stride BM
  v8f acc0 = {}, acc1 = {}, acc2 = {}, acc3 = {};
  for (int k0 = 0; k0 < MTOK; k0 += 32) {
    v16h a = load_a_frag(ab, MTOK, s0, k0, lane);
    v16h b0 = load_b_frag(vb, (size_t)BM, n0, k0, lane);
    v16h b1 = load_b_frag(vb, (size_t)BM, n0 + 16, k0, lane);
    v16h b2 = load_b_frag(vb, (size_t)BM, n0 + 32, k0, lane);
    v16h b3 = load_b_frag(vb, (size_t)BM, n0 + 48, k0, lane);
    acc0 = wmma_f16(a, b0, acc0);
    acc1 = wmma_f16(a, b1, acc1);
    acc2 = wmma_f16(a, b2, acc2);
    acc3 = wmma_f16(a, b3, acc3);
  }
  const int lr = lane & 15, hi = lane >> 4;
#pragma unroll
  for (int r = 0; r < 8; ++r) {
    const int slot = s0 + r + 8 * hi;
    const float inv = 1.0f / (rowsum[b * KSLOT + slot] + 1e-6f);
    size_t row = (size_t)(b * KSLOT + slot) * DD;
    upd[row + n0 + lr]      = (_Float16)(acc0[r] * inv);
    upd[row + n0 + 16 + lr] = (_Float16)(acc1[r] * inv);
    upd[row + n0 + 32 + lr] = (_Float16)(acc2[r] * inv);
    upd[row + n0 + 48 + lr] = (_Float16)(acc3[r] * inv);
  }
}

// ---------------------------------------------------------------------------
// Fused GRU cell: 6 WMMA accumulators (r/z/n for x@Wih^T and h@Whh^T) per
// 16x16 output tile, then the gate nonlinearity + state update in f32.
// ---------------------------------------------------------------------------
__global__ void __launch_bounds__(256)
gru_fused(const _Float16* __restrict__ upd16, const _Float16* __restrict__ h16,
          const _Float16* __restrict__ wih16, const _Float16* __restrict__ whh16,
          const float* __restrict__ bih, const float* __restrict__ bhh,
          const float* __restrict__ hOld, float* __restrict__ hNew) {
  const int lane = threadIdx.x & 31;
  const int wid = (blockIdx.x << 3) + (threadIdx.x >> 5);
  const int r0 = (wid >> 6) << 4;   // 64 col-tiles of 16
  const int c0 = (wid & 63) << 4;
  v8f xr = {}, xz = {}, xn = {}, hr = {}, hz = {}, hn = {};
  for (int k0 = 0; k0 < DD; k0 += 32) {
    v16h au = load_a_frag(upd16, DD, r0, k0, lane);
    v16h ah = load_a_frag(h16, DD, r0, k0, lane);
    xr = wmma_f16(au, load_b_frag(wih16, DD, c0, k0, lane), xr);
    xz = wmma_f16(au, load_b_frag(wih16, DD, c0 + DD, k0, lane), xz);
    xn = wmma_f16(au, load_b_frag(wih16, DD, c0 + 2 * DD, k0, lane), xn);
    hr = wmma_f16(ah, load_b_frag(whh16, DD, c0, k0, lane), hr);
    hz = wmma_f16(ah, load_b_frag(whh16, DD, c0 + DD, k0, lane), hz);
    hn = wmma_f16(ah, load_b_frag(whh16, DD, c0 + 2 * DD, k0, lane), hn);
  }
  const int lr = lane & 15, hi = lane >> 4;
  const int cn = c0 + lr;
  const float br = bih[cn],          bhr = bhh[cn];
  const float bz = bih[cn + DD],     bhz = bhh[cn + DD];
  const float bn = bih[cn + 2 * DD], bhn = bhh[cn + 2 * DD];
#pragma unroll
  for (int r = 0; r < 8; ++r) {
    const int row = r0 + r + 8 * hi;
    const float rg = 1.0f / (1.0f + __expf(-(xr[r] + br + hr[r] + bhr)));
    const float zg = 1.0f / (1.0f + __expf(-(xz[r] + bz + hz[r] + bhz)));
    const float ng = tanhf(xn[r] + bn + rg * (hn[r] + bhn));
    const float h = hOld[(size_t)row * DD + cn];
    hNew[(size_t)row * DD + cn] = (1.0f - zg) * ng + zg * h;
  }
}

// ---------------------------------------------------------------------------
extern "C" void kernel_launch(void* const* d_in, const int* in_sizes, int n_in,
                              void* d_out, int out_size, void* d_ws, size_t ws_size,
                              hipStream_t stream) {
  (void)in_sizes; (void)n_in; (void)out_size; (void)ws_size;
  const float* slots = (const float*)d_in[0];
  const float* P     = (const float*)d_in[1];
  const float* Wq    = (const float*)d_in[2];
  const float* Wk    = (const float*)d_in[3];
  const float* Wv    = (const float*)d_in[4];
  const float* wih   = (const float*)d_in[5];
  const float* whh   = (const float*)d_in[6];
  const float* bih   = (const float*)d_in[7];
  const float* bhh   = (const float*)d_in[8];
  const float* snw   = (const float*)d_in[9];
  const float* inw   = (const float*)d_in[10];
  float* out = (float*)d_out;

  char* wp = (char*)d_ws;
  auto take = [&](size_t bytes) -> char* {
    char* r = wp;
    wp += (bytes + 255) & ~(size_t)255;
    return r;
  };
  _Float16* Pn16  = (_Float16*)take((size_t)BM * DD * 2);   // 64 MB
  _Float16* k16   = (_Float16*)take((size_t)BM * DD * 2);   // 64 MB
  _Float16* vT    = (_Float16*)take((size_t)BM * DD * 2);   // 64 MB (d x B*M)
  _Float16* Wq16  = (_Float16*)take((size_t)DD * DD * 2);
  _Float16* Wk16  = (_Float16*)take((size_t)DD * DD * 2);
  _Float16* Wv16  = (_Float16*)take((size_t)DD * DD * 2);
  _Float16* wih16 = (_Float16*)take((size_t)3 * DD * DD * 2);
  _Float16* whh16 = (_Float16*)take((size_t)3 * DD * DD * 2);
  _Float16* q16   = (_Float16*)take((size_t)BKR * DD * 2);
  _Float16* sn16  = (_Float16*)take((size_t)BKR * DD * 2);
  _Float16* s16r  = (_Float16*)take((size_t)BKR * DD * 2);
  _Float16* attn  = (_Float16*)take((size_t)BB * KSLOT * MTOK * 2);  // 4 MB
  _Float16* upd16 = (_Float16*)take((size_t)BKR * DD * 2);
  float* rowsum   = (float*)take((size_t)BKR * 4);
  float* slotsA   = (float*)take((size_t)BKR * DD * 4);
  float* slotsB   = (float*)take((size_t)BKR * DD * 4);

  // --- one-time: weight converts, input rmsnorm, K/V projections ----------
  cvt_f32_to_f16<<<(DD * DD + 255) / 256, 256, 0, stream>>>(Wq, Wq16, DD * DD);
  cvt_f32_to_f16<<<(DD * DD + 255) / 256, 256, 0, stream>>>(Wk, Wk16, DD * DD);
  cvt_f32_to_f16<<<(DD * DD + 255) / 256, 256, 0, stream>>>(Wv, Wv16, DD * DD);
  cvt_f32_to_f16<<<(3 * DD * DD + 255) / 256, 256, 0, stream>>>(wih, wih16, 3 * DD * DD);
  cvt_f32_to_f16<<<(3 * DD * DD + 255) / 256, 256, 0, stream>>>(whh, whh16, 3 * DD * DD);

  rmsnorm_f16<<<BM, 256, 0, stream>>>(P, inw, Pn16, (_Float16*)nullptr, DD);

  // k16[BM x d] = Pn16 @ Wk16^T     (256 row-blocks * 16 strips)
  gemm_nt_f16<<<4096, 256, 0, stream>>>(Pn16, Wk16, k16, BM, DD, DD);
  // vT[d x BM]  = Wv16 @ Pn16^T     (8 row-blocks * 512 strips)
  gemm_nt_f16<<<4096, 256, 0, stream>>>(Wv16, Pn16, vT, DD, BM, DD);

  // --- 3 slot-attention iterations ----------------------------------------
  const float* cur = slots;
  for (int it = 0; it < 3; ++it) {
    float* dst = (it == 2) ? out : ((it & 1) ? slotsA : slotsB);
    zero_f32<<<1, 512, 0, stream>>>(rowsum, BKR);
    rmsnorm_f16<<<BKR, 256, 0, stream>>>(cur, snw, sn16, s16r, DD);
    // q16 = sn16 @ Wq16^T   (4 row-blocks * 16 strips)
    gemm_nt_f16<<<64, 256, 0, stream>>>(sn16, Wq16, q16, BKR, DD, DD);
    // scores + slot-softmax + partial M-sums (8*128 tiles -> 128 blocks)
    scores_softmax<<<128, 256, 0, stream>>>(q16, k16, attn, rowsum);
    // updates = (attn / (rowsum+eps)) @ V   (8*64 tiles -> 64 blocks)
    attn_v_gemm<<<64, 256, 0, stream>>>(attn, vT, rowsum, upd16);
    // fused GRU cell -> next slots (2048 tiles -> 256 blocks)
    gru_fused<<<256, 256, 0, stream>>>(upd16, s16r, wih16, whh16, bih, bhh,
                                       cur, dst);
    cur = dst;
  }
}
